// MyModule_63067299774675
// MI455X (gfx1250) — matrix operand order repaired and checked by reading, not compiled, not run on hardware
//
#include <hip/hip_runtime.h>
#include <hip/hip_bf16.h>
#include <math.h>

// ---------------------------------------------------------------------------
// Grid RNN: h[b,t,k] = tanh(W_ih[k] @ in + b_ih[k] + W_hh[k] @ h[b,t-1,k] + b_hh[k])
// bf16 WMMA (v_wmma_f32_16x16x32_bf16) for the per-layer input projection GEMM
// and the per-timestep recurrence GEMM. Register-blocked to minimize VMEM
// issue slots per WMMA (matrix pipe co-executes with loads on CDNA5).
// ---------------------------------------------------------------------------

typedef __bf16 bf16_t;
typedef __attribute__((ext_vector_type(16))) __bf16 v16bf;
typedef __attribute__((ext_vector_type(8)))  __bf16 v8bf;
typedef __attribute__((ext_vector_type(8)))  float  v8f;

#define BB   32
#define TT   512
#define DD   1024
#define HH   1024
#define NLAY 4

union frag16 { v16bf v; v8bf h[2]; };

// A-fragment (16x32 bf16, row-major source, ld in elements).
// Lanes 0-15: rows 0-15, K = {kb..kb+7, kb+16..kb+23};
// lanes 16-31: rows 0-15, K = {kb+8..kb+15, kb+24..kb+31}.
__device__ __forceinline__ v16bf load_a_frag(const bf16_t* __restrict__ base,
                                             int lane, int kb, int ld) {
    const int row = lane & 15;
    const int hi  = (lane >> 4) & 1;
    const bf16_t* p = base + (size_t)row * ld + kb + hi * 8;
    frag16 f;
    f.h[0] = *(const v8bf*)(p);
    f.h[1] = *(const v8bf*)(p + 16);
    return f.v;
}

// B-fragment (32x16 bf16). B column n == row n of W ([N,K] row-major, out=h@W^T).
// Lanes 0-15: K={kb..kb+7, kb+16..kb+23} of col n=lane; lanes 16-31: K={kb+8..kb+15, kb+24..kb+31} of col n=lane-16.
__device__ __forceinline__ v16bf load_b_frag(const bf16_t* __restrict__ wbase,
                                             int lane, int kb, int ld) {
    const int col = lane & 15;
    const int hi  = (lane >> 4) & 1;
    const bf16_t* p = wbase + (size_t)col * ld + kb + hi * 8;
    frag16 f;
    f.h[0] = *(const v8bf*)(p);
    f.h[1] = *(const v8bf*)(p + 16);
    return f.v;
}

#define WMMA_BF16(A_, B_, C_) \
    __builtin_amdgcn_wmma_f32_16x16x32_bf16(false, (A_), false, (B_), (short)0, (C_), false, false)

// ---------------------------------------------------------------------------
// fp32 -> bf16 conversion / zero (grid-stride)
// ---------------------------------------------------------------------------
__global__ void cvt_f32_bf16(const float* __restrict__ in, bf16_t* __restrict__ out, int n) {
    for (int i = blockIdx.x * blockDim.x + threadIdx.x; i < n; i += gridDim.x * blockDim.x)
        out[i] = (bf16_t)in[i];
}

__global__ void zero_bf16(bf16_t* __restrict__ p, int n) {
    for (int i = blockIdx.x * blockDim.x + threadIdx.x; i < n; i += gridDim.x * blockDim.x)
        p[i] = (bf16_t)0.0f;
}

// ---------------------------------------------------------------------------
// Input projection: XI[M,N] = A[M,K] @ W[N,K]^T + bias,  M=B*T=16384, K=D, N=H.
// Each wave computes a 64x32 block (4 M-tiles x 2 N-tiles): per K-chunk,
// 12 b128 loads feed 8 WMMAs (1.5 loads/wmma). 8 waves/block.
// Grid = (M/64 * N/32) / 8 = 1024 blocks.
// ---------------------------------------------------------------------------
__global__ __launch_bounds__(256) void xi_gemm(const bf16_t* __restrict__ A,
                                               const bf16_t* __restrict__ W,
                                               const float*  __restrict__ bias,
                                               float*        __restrict__ XI) {
    const int M = BB * TT, K = DD, N = HH;
    const int lane = threadIdx.x & 31;
    const int wave = threadIdx.x >> 5;
    const int g    = blockIdx.x * 8 + wave;
    const int NG   = N / 32;                 // 32 column groups
    const int mg   = g / NG;                 // 0..255, 64-row group
    const int ng   = g % NG;
    if (mg >= M / 64) return;                // uniform per wave; EXEC stays all-1s

    const bf16_t* A0  = A + ((size_t)mg * 64) * K;
    const bf16_t* Wb0 = W + ((size_t)ng * 32) * K;
    const bf16_t* Wb1 = Wb0 + (size_t)16 * K;

    v8f c[4][2] = {};
    for (int kb = 0; kb < K; kb += 32) {
        if (kb + 32 < K)   // gfx1250 global_prefetch_b8 path
            __builtin_prefetch(A0 + (size_t)(lane & 15) * K + kb + 32, 0, 3);
        v16bf b0 = load_b_frag(Wb0, lane, kb, K);
        v16bf b1 = load_b_frag(Wb1, lane, kb, K);
        v16bf a0 = load_a_frag(A0 + (size_t) 0 * K, lane, kb, K);
        v16bf a1 = load_a_frag(A0 + (size_t)16 * K, lane, kb, K);
        v16bf a2 = load_a_frag(A0 + (size_t)32 * K, lane, kb, K);
        v16bf a3 = load_a_frag(A0 + (size_t)48 * K, lane, kb, K);
        c[0][0] = WMMA_BF16(a0, b0, c[0][0]);
        c[0][1] = WMMA_BF16(a0, b1, c[0][1]);
        c[1][0] = WMMA_BF16(a1, b0, c[1][0]);
        c[1][1] = WMMA_BF16(a1, b1, c[1][1]);
        c[2][0] = WMMA_BF16(a2, b0, c[2][0]);
        c[2][1] = WMMA_BF16(a2, b1, c[2][1]);
        c[3][0] = WMMA_BF16(a3, b0, c[3][0]);
        c[3][1] = WMMA_BF16(a3, b1, c[3][1]);
    }

    const int col0 = ng * 32 + (lane & 15);
    const int rlo  = (lane >> 4) ? 8 : 0;
    const float bi0 = bias[col0], bi1 = bias[col0 + 16];
#pragma unroll
    for (int i = 0; i < 4; ++i) {
#pragma unroll
        for (int r = 0; r < 8; ++r) {
            const int row = mg * 64 + i * 16 + rlo + r;
            float* xrow = XI + (size_t)row * N;
            xrow[col0 +  0] = c[i][0][r] + bi0;
            xrow[col0 + 16] = c[i][1][r] + bi1;
        }
    }
}

// ---------------------------------------------------------------------------
// One recurrence step: h_t = tanh(XI[:,t] + h_{t-1} @ Whh^T + b_hh).
// Each wave computes both M-tiles (M=32) for one N-tile, sharing the weight
// fragment across 2 WMMAs: 6 loads / 2 wmma per K-chunk, and each W_hh
// fragment is pulled from L2 exactly once per step.
// 64 waves -> 8 blocks of 8 waves. Double-buffered h state avoids intra-step
// RAW races across blocks.
// ---------------------------------------------------------------------------
__global__ __launch_bounds__(256) void rnn_step(const bf16_t* __restrict__ hprev,   // [B,H]
                                                const bf16_t* __restrict__ Whh,     // [H,H]
                                                const float*  __restrict__ XI,      // [B*T,H]
                                                const float*  __restrict__ bhh,     // [H]
                                                const int*    __restrict__ seq_lens,// [B]
                                                bf16_t*       __restrict__ hnext,   // [B,H]
                                                bf16_t*       __restrict__ out_bf,  // [B*T,H]
                                                float*        __restrict__ out,     // [B,T,NLAY,H]
                                                int t, int layer) {
    const int lane  = threadIdx.x & 31;
    const int wave  = threadIdx.x >> 5;
    const int ntile = blockIdx.x * 8 + wave;     // 0..63

    const bf16_t* A0    = hprev;                       // batch rows 0..15
    const bf16_t* A1    = hprev + (size_t)16 * HH;     // batch rows 16..31
    const bf16_t* Wbase = Whh + (size_t)ntile * 16 * HH;

    v8f c0 = {}, c1 = {};
    for (int kb = 0; kb < HH; kb += 32) {
        v16bf b  = load_b_frag(Wbase, lane, kb, HH);
        v16bf a0 = load_a_frag(A0, lane, kb, HH);
        v16bf a1 = load_a_frag(A1, lane, kb, HH);
        c0 = WMMA_BF16(a0, b, c0);
        c1 = WMMA_BF16(a1, b, c1);
    }

    const int col = ntile * 16 + (lane & 15);
    const int rlo = (lane >> 4) ? 8 : 0;
    const float bb = bhh[col];
#pragma unroll
    for (int r = 0; r < 8; ++r) {
        // M-tile 0: batch rows 0..15
        {
            const int b = rlo + r;
            const size_t bt = (size_t)b * TT + t;
            const float v = tanhf(c0[r] + XI[bt * HH + col] + bb);
            hnext[(size_t)b * HH + col] = (bf16_t)v;
            out_bf[bt * HH + col]       = (bf16_t)v;
            const float m = (t < seq_lens[b]) ? 1.0f : 0.0f;
            out[(bt * NLAY + layer) * HH + col] = v * m;
        }
        // M-tile 1: batch rows 16..31
        {
            const int b = 16 + rlo + r;
            const size_t bt = (size_t)b * TT + t;
            const float v = tanhf(c1[r] + XI[bt * HH + col] + bb);
            hnext[(size_t)b * HH + col] = (bf16_t)v;
            out_bf[bt * HH + col]       = (bf16_t)v;
            const float m = (t < seq_lens[b]) ? 1.0f : 0.0f;
            out[(bt * NLAY + layer) * HH + col] = v * m;
        }
    }
}

// ---------------------------------------------------------------------------
// Host driver
// ---------------------------------------------------------------------------
extern "C" void kernel_launch(void* const* d_in, const int* in_sizes, int n_in,
                              void* d_out, int out_size, void* d_ws, size_t ws_size,
                              hipStream_t stream) {
    const float* x        = (const float*)d_in[0];  // [B,T,D]
    const int*   seq_lens = (const int*)  d_in[1];  // [B]
    const float* w_ih     = (const float*)d_in[2];  // [NLAY,H,D]
    const float* w_hh     = (const float*)d_in[3];  // [NLAY,H,H]
    const float* b_ih     = (const float*)d_in[4];  // [NLAY,H]
    const float* b_hh     = (const float*)d_in[5];  // [NLAY,H]
    float* out = (float*)d_out;                     // [B,T,NLAY,H]

    // Workspace carve-up (256B aligned slices)
    char* ws = (char*)d_ws;
    size_t off = 0;
    auto carve = [&](size_t bytes) -> void* {
        void* p = ws + off;
        off += (bytes + 255) & ~(size_t)255;
        return p;
    };
    bf16_t* wih_bf = (bf16_t*)carve((size_t)NLAY * HH * DD * sizeof(bf16_t)); // 8 MB
    bf16_t* whh_bf = (bf16_t*)carve((size_t)NLAY * HH * HH * sizeof(bf16_t)); // 8 MB
    bf16_t* buf0   = (bf16_t*)carve((size_t)BB * TT * HH * sizeof(bf16_t));   // 32 MB
    bf16_t* buf1   = (bf16_t*)carve((size_t)BB * TT * HH * sizeof(bf16_t));   // 32 MB
    float*  XI     = (float*) carve((size_t)BB * TT * HH * sizeof(float));    // 64 MB
    bf16_t* hbuf   = (bf16_t*)carve((size_t)2 * BB * HH * sizeof(bf16_t));    // 128 KB
    (void)ws_size; (void)in_sizes; (void)n_in; (void)out_size;

    // Precision conversion: weights + layer-0 input to bf16
    {
        const int nW = NLAY * HH * DD;
        cvt_f32_bf16<<<nW / 256, 256, 0, stream>>>(w_ih, wih_bf, nW);
        cvt_f32_bf16<<<nW / 256, 256, 0, stream>>>(w_hh, whh_bf, nW);
        const int nX = BB * TT * DD;
        cvt_f32_bf16<<<nX / 256, 256, 0, stream>>>(x, buf0, nX);
    }

    const int xi_blocks   = (BB * TT / 64) * (HH / 32) / 8; // 1024
    const int step_blocks = (HH / 16) / 8;                  // 8

    for (int k = 0; k < NLAY; ++k) {
        bf16_t* in_bf  = (k & 1) ? buf1 : buf0;
        bf16_t* out_bf = (k & 1) ? buf0 : buf1;

        // Hoisted input projection: one big WMMA GEMM per layer
        xi_gemm<<<xi_blocks, 256, 0, stream>>>(in_bf,
                                               wih_bf + (size_t)k * HH * DD,
                                               b_ih + (size_t)k * HH, XI);

        // h0 = 0 (zero both ping-pong h buffers)
        zero_bf16<<<(2 * BB * HH) / 256, 256, 0, stream>>>(hbuf, 2 * BB * HH);

        // Sequential recurrence over T; double-buffered hidden state
        for (int t = 0; t < TT; ++t) {
            rnn_step<<<step_blocks, 256, 0, stream>>>(
                hbuf + (size_t)(t & 1) * BB * HH,
                whh_bf + (size_t)k * HH * HH,
                XI,
                b_hh + (size_t)k * HH,
                seq_lens,
                hbuf + (size_t)((t + 1) & 1) * BB * HH,
                out_bf,
                out,
                t, k);
        }
    }
}
